// CausalSelfAttention_51359218925814
// MI455X (gfx1250) — compile-verified, matrix operand order checked by reading, not run on hardware
//
#include <hip/hip_runtime.h>

typedef _Float16 h16;
typedef __attribute__((ext_vector_type(16))) _Float16 v16h;
typedef __attribute__((ext_vector_type(8)))  float    v8f;
typedef __attribute__((ext_vector_type(4)))  unsigned int v4u;

union F16Frag { v16h v; v4u u[2]; };

__device__ __forceinline__ v8f v8f_zero() {
  v8f z;
#pragma unroll
  for (int i = 0; i < 8; ++i) z[i] = 0.0f;
  return z;
}

__device__ __forceinline__ v8f wmma_f16(v16h a, v16h b, v8f c) {
  // D = A(16x32 f16) * B(32x16 f16) + C(16x16 f32)
  return __builtin_amdgcn_wmma_f32_16x16x32_f16(false, a, false, b, (short)0, c, false, false);
}

// Async global->LDS copy of 32 bytes per lane (ASYNCcnt path, gfx1250).
// offset: immediate is added to BOTH the global address and the LDS address,
// so off / offset:16 moves a contiguous 32B segment.
__device__ __forceinline__ void async_copy_b32bytes(unsigned lds_off, const void* gaddr) {
  asm volatile("global_load_async_to_lds_b128 %0, %1, off\n\t"
               "global_load_async_to_lds_b128 %0, %1, off offset:16"
               :: "v"(lds_off), "v"(gaddr) : "memory");
}

__device__ __forceinline__ void wait_asynccnt0() {
  asm volatile("s_wait_asynccnt 0x0" ::: "memory");
}

static constexpr int B   = 4;
static constexpr int T   = 1024;
static constexpr int DIM = 2048;
static constexpr int H   = 16;
static constexpr int KVH = 4;
static constexpr int HD  = 128;
static constexpr int G   = H / KVH;          // 4
static constexpr int NKV = 2 * KVH * HD;     // 1024

// ---------------- elementwise / layout kernels ----------------

__global__ void f32_to_f16_kernel(const float* __restrict__ in, h16* __restrict__ out, int n) {
  int i = blockIdx.x * blockDim.x + threadIdx.x;
  if (i < n) out[i] = (h16)in[i];
}

// W is (K x N) row-major f32; WT becomes (N x K) row-major f16.
__global__ void transpose_to_f16_kernel(const float* __restrict__ W, h16* __restrict__ WT,
                                        int K, int N) {
  int i = blockIdx.x * blockDim.x + threadIdx.x;
  if (i >= K * N) return;
  int k = i / N, n = i % N;
  WT[(size_t)n * K + k] = (h16)W[i];
}

// In-place RoPE on f16 buffer laid out [(b*T+t)*row_stride + h*HD + d].
__global__ void rope_kernel(h16* __restrict__ u, const float* __restrict__ cs,
                            const float* __restrict__ sn, int heads, int row_stride) {
  int i = blockIdx.x * blockDim.x + threadIdx.x;
  int total = B * T * heads * (HD / 2);
  if (i >= total) return;
  int d  = i % (HD / 2);
  int h  = (i / (HD / 2)) % heads;
  int bt = i / ((HD / 2) * heads);
  int t  = bt % T;
  size_t base = (size_t)bt * row_stride + (size_t)h * HD;
  float u1 = (float)u[base + d];
  float u2 = (float)u[base + 64 + d];
  float c = cs[t * 64 + d];
  float s = sn[t * 64 + d];
  u[base + d]      = (h16)(u1 * c - u2 * s);
  u[base + 64 + d] = (h16)(u1 * s + u2 * c);
}

// kv layout [(b*T+t)*NKV + {0:k,1:v}*KVH*HD + h*HD + d]  ->  vt[b][h][d][t]
__global__ void transpose_v_kernel(const h16* __restrict__ kv, h16* __restrict__ vt, int n) {
  int i = blockIdx.x * blockDim.x + threadIdx.x;
  if (i >= n) return;
  int d  = i % HD;
  int h  = (i / HD) % KVH;
  int bt = i / (HD * KVH);
  int b  = bt / T;
  int t  = bt % T;
  vt[(((size_t)b * KVH + h) * HD + d) * T + t] =
      kv[(size_t)bt * NKV + (size_t)KVH * HD + (size_t)h * HD + d];
}

// ---------------- f16 WMMA GEMM: C[M,N] = A[M,K] * BT[N,K]^T ----------------
// 256 threads = 8 waves arranged 2(M) x 4(N); 128x128 tile; K-step 32.
// Tile staging uses the gfx1250 async global->LDS copy engine (ASYNCcnt).

__global__ __launch_bounds__(256) void gemm_f16_kernel(
    const h16* __restrict__ A, const h16* __restrict__ BT,
    float* __restrict__ Cf, h16* __restrict__ Ch, int M, int N, int K) {
  __shared__ h16 sA[128 * 32];
  __shared__ h16 sB[128 * 32];

  const int tid    = threadIdx.x;
  const int lane   = tid & 31;
  const int wave   = tid >> 5;
  const int wm     = wave >> 2;          // 0..1
  const int wn     = wave & 3;           // 0..3
  const int hgrp   = (lane >> 4) & 1;    // lane half (ISA 16-bit A/B layout)
  const int ln     = lane & 15;
  const int tilesN = N >> 7;
  const int m0 = (blockIdx.x / tilesN) << 7;
  const int n0 = (blockIdx.x % tilesN) << 7;

  const int lrow = tid >> 1;             // 0..127
  const int lseg = (tid & 1) << 4;       // 0 or 16 halves

  v8f acc[4][2];
#pragma unroll
  for (int mt = 0; mt < 4; ++mt)
#pragma unroll
    for (int nt = 0; nt < 2; ++nt) acc[mt][nt] = v8f_zero();

  const h16* Ap = A  + (size_t)(m0 + lrow) * K + lseg;
  const h16* Bp = BT + (size_t)(n0 + lrow) * K + lseg;

  // Workgroup-relative LDS byte offsets (generic LDS pointers carry the
  // LDS offset in their low 32 bits on amdgcn).
  const unsigned ldsA = (unsigned)(uintptr_t)(sA + lrow * 32 + lseg);
  const unsigned ldsB = (unsigned)(uintptr_t)(sB + lrow * 32 + lseg);

  for (int kk = 0; kk < K; kk += 32) {
    __syncthreads();                       // previous iteration's reads done
    async_copy_b32bytes(ldsA, Ap + kk);    // global_load_async_to_lds_b128 x2
    async_copy_b32bytes(ldsB, Bp + kk);
    if (kk + 32 < K) {
      __builtin_prefetch(Ap + kk + 32, 0, 1);  // global_prefetch_b8
      __builtin_prefetch(Bp + kk + 32, 0, 1);
    }
    wait_asynccnt0();                      // s_wait_asynccnt 0
    __syncthreads();                       // tiles visible to all waves

    v16h afrag[4], bfrag[2];
#pragma unroll
    for (int mt = 0; mt < 4; ++mt) {
      const h16* p = sA + (wm * 64 + mt * 16 + ln) * 32;
      F16Frag f;
      f.u[0] = *(const v4u*)(p + 8 * hgrp);
      f.u[1] = *(const v4u*)(p + 16 + 8 * hgrp);
      afrag[mt] = f.v;
    }
#pragma unroll
    for (int nt = 0; nt < 2; ++nt) {
      const h16* p = sB + (wn * 32 + nt * 16 + ln) * 32;
      F16Frag f;
      f.u[0] = *(const v4u*)(p + 8 * hgrp);
      f.u[1] = *(const v4u*)(p + 16 + 8 * hgrp);
      bfrag[nt] = f.v;
    }
#pragma unroll
    for (int mt = 0; mt < 4; ++mt)
#pragma unroll
      for (int nt = 0; nt < 2; ++nt)
        acc[mt][nt] = wmma_f16(afrag[mt], bfrag[nt], acc[mt][nt]);
  }

  // Epilogue. C frag: lane halves hold rows r / r+8, lane%16 = column.
#pragma unroll
  for (int mt = 0; mt < 4; ++mt) {
#pragma unroll
    for (int nt = 0; nt < 2; ++nt) {
#pragma unroll
      for (int r = 0; r < 8; ++r) {
        int m = m0 + wm * 64 + mt * 16 + r + 8 * hgrp;
        int n = n0 + wn * 32 + nt * 16 + ln;
        size_t idx = (size_t)m * N + n;
        if (Ch) Ch[idx] = (h16)acc[mt][nt][r];
        else    Cf[idx] = acc[mt][nt][r];
      }
    }
  }
}

// ---------------- causal flash attention (per-wave, 16 q rows / wave) ----------------

__global__ __launch_bounds__(256) void attn_kernel(
    const h16* __restrict__ qh,   // (B,T,H,HD) f16, RoPE applied
    const h16* __restrict__ kv,   // (B,T,2,KVH,HD) f16, RoPE applied to k
    const h16* __restrict__ vt,   // (B,KVH,HD,T) f16
    h16* __restrict__ ah) {       // (B,T,H,HD) f16
  __shared__ h16 ldsP[8 * 16 * 32];   // per-wave 16x32 P bounce buffer

  const int lane = threadIdx.x & 31;
  const int wave = threadIdx.x >> 5;
  const int hgrp = (lane >> 4) & 1;
  const int ln   = lane & 15;

  int bid = blockIdx.x;
  const int qtiles = T / 128;
  const int qt = bid % qtiles; bid /= qtiles;
  const int h  = bid % H;      bid /= H;
  const int b  = bid;
  const int hk = h / G;
  const int qbase = qt * 128 + wave * 16;
  h16* lp = ldsP + wave * (16 * 32);

  // Q fragments: 16 rows x 128 (4 K-steps of 32)
  v16h qf[4];
  {
    const h16* qrow = qh + ((size_t)(b * T + qbase + ln) * H + h) * HD;
#pragma unroll
    for (int kc = 0; kc < 4; ++kc) {
      F16Frag f;
      f.u[0] = *(const v4u*)(qrow + kc * 32 + 8 * hgrp);
      f.u[1] = *(const v4u*)(qrow + kc * 32 + 16 + 8 * hgrp);
      qf[kc] = f.v;
    }
  }

  float mrow[8], lsum[8];
  v8f o[8];
#pragma unroll
  for (int r = 0; r < 8; ++r) { mrow[r] = -3.0e38f; lsum[r] = 0.0f; }
#pragma unroll
  for (int dc = 0; dc < 8; ++dc) o[dc] = v8f_zero();

  const float scale = 0.08838834764831845f;  // 1/sqrt(128)

  for (int kb = 0; kb <= qbase; kb += 32) {
    // S = Q * K^T for keys [kb,kb+16) and [kb+16,kb+32)
    v8f s0 = v8f_zero(), s1 = v8f_zero();
#pragma unroll
    for (int sub = 0; sub < 2; ++sub) {
      const int key = kb + sub * 16 + ln;
      const h16* krow = kv + ((size_t)(b * T + key) * 2 + 0) * (KVH * HD) + (size_t)hk * HD;
      v8f sacc = v8f_zero();
#pragma unroll
      for (int kc = 0; kc < 4; ++kc) {
        F16Frag f;
        f.u[0] = *(const v4u*)(krow + kc * 32 + 8 * hgrp);
        f.u[1] = *(const v4u*)(krow + kc * 32 + 16 + 8 * hgrp);
        sacc = wmma_f16(qf[kc], f.v, sacc);
      }
      if (sub == 0) s0 = sacc; else s1 = sacc;
    }

    const bool needMask = (kb + 31) > qbase;
#pragma unroll
    for (int r = 0; r < 8; ++r) {
      const int grow = qbase + r + 8 * hgrp;
      float a0 = s0[r] * scale;
      float a1 = s1[r] * scale;
      if (needMask) {
        if (kb + ln > grow)      a0 = -3.0e38f;
        if (kb + 16 + ln > grow) a1 = -3.0e38f;
      }
      float rm = fmaxf(a0, a1);
#pragma unroll
      for (int mm = 1; mm < 16; mm <<= 1) rm = fmaxf(rm, __shfl_xor(rm, mm, 32));
      const float mnew = fmaxf(mrow[r], rm);
      const float corr = __expf(mrow[r] - mnew);
      const float p0 = __expf(a0 - mnew);
      const float p1 = __expf(a1 - mnew);
      float ps = p0 + p1;
#pragma unroll
      for (int mm = 1; mm < 16; mm <<= 1) ps += __shfl_xor(ps, mm, 32);
      lsum[r] = lsum[r] * corr + ps;
      mrow[r] = mnew;
#pragma unroll
      for (int dc = 0; dc < 8; ++dc) o[dc][r] *= corr;
      // stash P (C-layout -> row-major 16x32 in LDS)
      lp[(r + 8 * hgrp) * 32 + ln]      = (h16)p0;
      lp[(r + 8 * hgrp) * 32 + 16 + ln] = (h16)p1;
    }

    // Re-read P as an A fragment (row in lane, K along keys)
    F16Frag pf;
    pf.u[0] = *(const v4u*)(lp + ln * 32 + 8 * hgrp);
    pf.u[1] = *(const v4u*)(lp + ln * 32 + 16 + 8 * hgrp);

    // O += P * V   (V pre-transposed: contiguous along t)
#pragma unroll
    for (int dc = 0; dc < 8; ++dc) {
      const h16* vrow = vt + (((size_t)b * KVH + hk) * HD + dc * 16 + ln) * T + kb;
      F16Frag vf;
      vf.u[0] = *(const v4u*)(vrow + 8 * hgrp);
      vf.u[1] = *(const v4u*)(vrow + 16 + 8 * hgrp);
      o[dc] = wmma_f16(pf.v, vf.v, o[dc]);
    }
  }

  // Normalize and write attention output (f16) feeding the Wo GEMM.
#pragma unroll
  for (int dc = 0; dc < 8; ++dc) {
#pragma unroll
    for (int r = 0; r < 8; ++r) {
      const int t = qbase + r + 8 * hgrp;
      const int d = dc * 16 + ln;
      ah[(size_t)(b * T + t) * DIM + (size_t)h * HD + d] = (h16)(o[dc][r] / lsum[r]);
    }
  }
}

// ---------------- host orchestration ----------------

extern "C" void kernel_launch(void* const* d_in, const int* in_sizes, int n_in,
                              void* d_out, int out_size, void* d_ws, size_t ws_size,
                              hipStream_t stream) {
  (void)in_sizes; (void)n_in; (void)out_size; (void)ws_size;
  const float* x   = (const float*)d_in[0];
  const float* Wq  = (const float*)d_in[1];
  const float* Wkv = (const float*)d_in[2];
  const float* Wo  = (const float*)d_in[3];
  const float* cs  = (const float*)d_in[4];
  const float* sn  = (const float*)d_in[5];
  float* out = (float*)d_out;

  char* ws = (char*)d_ws;
  size_t off = 0;
  auto take = [&](size_t nhalf) -> h16* {
    h16* p = (h16*)(ws + off);
    off += (nhalf * sizeof(h16) + 255) & ~(size_t)255;
    return p;
  };
  h16* xah  = take((size_t)B * T * DIM);        // x in f16; reused as attention output
  h16* WqT  = take((size_t)DIM * DIM);
  h16* WkvT = take((size_t)DIM * NKV);
  h16* WoT  = take((size_t)DIM * DIM);
  h16* qh   = take((size_t)B * T * DIM);
  h16* kvh  = take((size_t)B * T * NKV);
  h16* vth  = take((size_t)B * KVH * HD * T);

  int n;
  n = B * T * DIM;
  f32_to_f16_kernel<<<(n + 255) / 256, 256, 0, stream>>>(x, xah, n);
  n = DIM * DIM;
  transpose_to_f16_kernel<<<(n + 255) / 256, 256, 0, stream>>>(Wq, WqT, DIM, DIM);
  n = DIM * NKV;
  transpose_to_f16_kernel<<<(n + 255) / 256, 256, 0, stream>>>(Wkv, WkvT, DIM, NKV);
  n = DIM * DIM;
  transpose_to_f16_kernel<<<(n + 255) / 256, 256, 0, stream>>>(Wo, WoT, DIM, DIM);

  // q = x @ Wq   (M=4096, N=2048, K=2048) -> f16
  gemm_f16_kernel<<<(B * T / 128) * (DIM / 128), 256, 0, stream>>>(
      xah, WqT, nullptr, qh, B * T, DIM, DIM);
  // kv = x @ Wkv (M=4096, N=1024, K=2048) -> f16
  gemm_f16_kernel<<<(B * T / 128) * (NKV / 128), 256, 0, stream>>>(
      xah, WkvT, nullptr, kvh, B * T, NKV, DIM);

  n = B * T * H * (HD / 2);
  rope_kernel<<<(n + 255) / 256, 256, 0, stream>>>(qh, cs, sn, H, DIM);
  n = B * T * KVH * (HD / 2);
  rope_kernel<<<(n + 255) / 256, 256, 0, stream>>>(kvh, cs, sn, KVH, NKV);
  n = B * T * KVH * HD;
  transpose_v_kernel<<<(n + 255) / 256, 256, 0, stream>>>(kvh, vth, n);

  attn_kernel<<<B * H * (T / 128), 256, 0, stream>>>(qh, kvh, vth, xah);

  // out = attn_out @ Wo (f32 output)
  gemm_f16_kernel<<<(B * T / 128) * (DIM / 128), 256, 0, stream>>>(
      xah, WoT, out, nullptr, B * T, DIM, DIM);
}
